// SumLayer_64003602645440
// MI455X (gfx1250) — matrix operand order repaired and checked by reading, not compiled, not run on hardware
//
#include <hip/hip_runtime.h>

typedef __attribute__((ext_vector_type(16))) _Float16 v16h;
typedef __attribute__((ext_vector_type(8)))  _Float16 v8h;
typedef __attribute__((ext_vector_type(8)))  float    v8f;
typedef __attribute__((ext_vector_type(4)))  float    v4f;

#define B_DIM 128
#define R_DIM 512
#define N_IN  64
#define N_SUM 64

#define L2E 1.4426950408889634f   // log2(e)
#define LN2 0.6931471805599453f   // ln(2)

__global__ __launch_bounds__(256)
void sumlayer_wmma_kernel(const float* __restrict__ inp,
                          const float* __restrict__ kern,
                          float* __restrict__ out) {
  // LDS: softmax probs P[s][i] (f16, row-major), per-wave E tiles, row maxes.
  __shared__ _Float16 sP[N_SUM * N_IN];    // 8 KB
  __shared__ _Float16 sE[8][16 * N_IN];    // 16 KB (one 16x64 tile per wave)
  __shared__ float    sM[B_DIM];           // 512 B

  const int r    = blockIdx.x;
  const int t    = threadIdx.x;
  const int lane = t & 31;
  const int wave = t >> 5;

  // ---------------- Stage 1: P[s][:] = softmax(kern[r][s][:]) -> f16 ----------------
  {
    const int s = t >> 2;        // 0..63 : one row per 4 threads
    const int q = t & 3;         // quarter (16 f32 each)
    const float* krow = kern + (((size_t)r * N_SUM + s) * N_IN) + q * 16;
    float x[16];
    #pragma unroll
    for (int i = 0; i < 4; ++i)
      *(v4f*)(x + 4 * i) = *(const v4f*)(krow + 4 * i);

    float mx = x[0];
    #pragma unroll
    for (int i = 1; i < 16; ++i) mx = fmaxf(mx, x[i]);
    mx = fmaxf(mx, __shfl_xor(mx, 1, 4));
    mx = fmaxf(mx, __shfl_xor(mx, 2, 4));

    // e = exp2((x - mx)*log2e) ; args <= 0 so raw v_exp_f32 is safe.
    const float nm = -mx * L2E;
    float e[16];
    float se = 0.0f;
    #pragma unroll
    for (int i = 0; i < 16; ++i) {
      e[i] = __builtin_amdgcn_exp2f(fmaf(x[i], L2E, nm));
      se += e[i];
    }
    se += __shfl_xor(se, 1, 4);
    se += __shfl_xor(se, 2, 4);
    const float inv = 1.0f / se;

    v8h h0, h1;
    #pragma unroll
    for (int i = 0; i < 8; ++i) {
      h0[i] = (_Float16)(e[i] * inv);
      h1[i] = (_Float16)(e[i + 8] * inv);
    }
    _Float16* dst = &sP[s * N_IN + q * 16];
    *(v8h*)(dst)     = h0;
    *(v8h*)(dst + 8) = h1;
  }
  __syncthreads();

  // ---------------- Stage 2: per-wave 16x64 E tile, then WMMA ----------------
  const int half = lane >> 4;          // 0 or 1
  const int row  = lane & 15;          // M row within tile / N col within tile
  const int brow0 = wave * 16;         // this wave's M-tile base (B rows)
  {
    const int b    = brow0 + row;
    const int colb = half * 32;        // each lane loads half a row (32 f32)
    const float* irow = inp + (((size_t)b * R_DIM + r) * N_IN) + colb;
    float x[32];
    #pragma unroll
    for (int i = 0; i < 8; ++i)
      *(v4f*)(x + 4 * i) = *(const v4f*)(irow + 4 * i);

    float mx = x[0];
    #pragma unroll
    for (int i = 1; i < 32; ++i) mx = fmaxf(mx, x[i]);
    mx = fmaxf(mx, __shfl_xor(mx, 16, 32));   // combine the two half-rows
    if (lane < 16) sM[b] = mx;

    const float nm = -mx * L2E;
    _Float16* erow = &sE[wave][row * N_IN + colb];
    #pragma unroll
    for (int i = 0; i < 4; ++i) {
      v8h h;
      #pragma unroll
      for (int j = 0; j < 8; ++j)
        h[j] = (_Float16)__builtin_amdgcn_exp2f(fmaf(x[8 * i + j], L2E, nm));
      *(v8h*)(erow + 8 * i) = h;
    }
  }
  __syncthreads();   // make E tiles + sM visible (and compiler-order LDS ops)

  // A fragments (ISA 16-bit A 16x32 layout): lane -> row M=lane%16,
  // halves a[0..7] = K = kk + half*8 + 0..7 ; a[8..15] = K = kk+16+half*8 + 0..7
  v16h a0, a1;
  {
    const _Float16* er = &sE[wave][row * N_IN];
    v8h lo0 = *(const v8h*)(er + 0  + half * 8);
    v8h hi0 = *(const v8h*)(er + 16 + half * 8);
    v8h lo1 = *(const v8h*)(er + 32 + half * 8);
    v8h hi1 = *(const v8h*)(er + 48 + half * 8);
    a0 = __builtin_shufflevector(lo0, hi0, 0,1,2,3,4,5,6,7,8,9,10,11,12,13,14,15);
    a1 = __builtin_shufflevector(lo1, hi1, 0,1,2,3,4,5,6,7,8,9,10,11,12,13,14,15);
  }

  float mrow[8];
  #pragma unroll
  for (int v = 0; v < 8; ++v) mrow[v] = sM[brow0 + half * 8 + v];

  // N tiles: out[b, r, nt*16 + row]
  #pragma unroll
  for (int nt = 0; nt < 4; ++nt) {
    // B fragments (K x N): lane holds column N=row of P (row-major), halves
    // b[0..7] = K = kk + half*8 + 0..7 ; b[8..15] = K = kk+16+half*8 + 0..7 (matches A).
    const _Float16* pr = &sP[(nt * 16 + row) * N_IN + half * 8];
    v8h blo0 = *(const v8h*)(pr + 0);
    v8h bhi0 = *(const v8h*)(pr + 16);
    v8h blo1 = *(const v8h*)(pr + 32);
    v8h bhi1 = *(const v8h*)(pr + 48);
    v16h b0 = __builtin_shufflevector(blo0, bhi0, 0,1,2,3,4,5,6,7,8,9,10,11,12,13,14,15);
    v16h b1 = __builtin_shufflevector(blo1, bhi1, 0,1,2,3,4,5,6,7,8,9,10,11,12,13,14,15);

    v8f c = {};
    c = __builtin_amdgcn_wmma_f32_16x16x32_f16(false, a0, false, b0,
                                               (short)0, c, false, false);
    c = __builtin_amdgcn_wmma_f32_16x16x32_f16(false, a1, false, b1,
                                               (short)0, c, false, false);

    // Epilogue: out[brow, r, s] = ln2*log2(c) + m[brow]
    // acc in [~5e-3, 1] -> never denormal/nonpositive, raw v_log_f32 is safe.
    #pragma unroll
    for (int v = 0; v < 8; ++v) {
      const int brow = brow0 + half * 8 + v;                 // C/D: M = v + 8*half
      const size_t o = ((size_t)brow * R_DIM + r) * N_SUM + nt * 16 + row;
      out[o] = fmaf(__builtin_amdgcn_logf(c[v]), LN2, mrow[v]);
    }
  }
}

extern "C" void kernel_launch(void* const* d_in, const int* in_sizes, int n_in,
                              void* d_out, int out_size, void* d_ws, size_t ws_size,
                              hipStream_t stream) {
  const float* inp  = (const float*)d_in[0];   // (128, 512, 64) f32
  const float* kern = (const float*)d_in[1];   // (512, 64, 64) f32
  float* out = (float*)d_out;                  // (128, 512, 64) f32
  (void)in_sizes; (void)n_in; (void)out_size; (void)d_ws; (void)ws_size;
  sumlayer_wmma_kernel<<<dim3(R_DIM), dim3(256), 0, stream>>>(inp, kern, out);
}
